// MultiHeadAttention_31061203485179
// MI455X (gfx1250) — matrix operand order repaired and checked by reading, not compiled, not run on hardware
//
#include <hip/hip_runtime.h>

typedef __attribute__((ext_vector_type(16))) __bf16 v16bf;
typedef __attribute__((ext_vector_type(8)))  float  v8f;
typedef __attribute__((ext_vector_type(4)))  int    v4i;

typedef __attribute__((address_space(1))) v4i gbl_v4i;
typedef __attribute__((address_space(3))) v4i lds_v4i;

namespace mha {
constexpr int B_  = 4;
constexpr int S_  = 2048;
constexpr int D_  = 1024;
constexpr int H_  = 16;
constexpr int DK_ = 64;
constexpr int M_  = B_ * S_;   // 8192 rows of x

constexpr int BK  = 32;        // k-chunk depth
constexpr int WGM = 64;        // rows per workgroup (16 per wave, 4 waves)
constexpr int WGN = 128;       // cols per workgroup
constexpr int LDB = WGN + 8;   // padded LDS row stride (elems): 272B -> conflict-free
}

#if defined(__AMDGCN__) && __has_builtin(__builtin_amdgcn_global_load_async_to_lds_b128) && \
    __has_builtin(__builtin_amdgcn_s_wait_asynccnt)
#define MHA_ASYNC_LDS 1
#else
#define MHA_ASYNC_LDS 0
#endif

__device__ __forceinline__ unsigned short f2bf(float f) {
  unsigned int u = __float_as_uint(f);
  u += 0x7fffu + ((u >> 16) & 1u);     // round-to-nearest-even
  return (unsigned short)(u >> 16);
}

union ABPack { uint4 u[2]; v16bf v; };

// A operand, 16x32 bf16, row-major source (stride ld elems).
// ISA layout: lane&15 = M row; lane>>4 selects K chunks {hl*8.., hl*8+16..}.
__device__ __forceinline__ v16bf load_a16x32(const unsigned short* base, int ld, int k0, int lane) {
  const int hl = lane >> 4, r = lane & 15;
  const unsigned short* p = base + (size_t)r * ld + k0 + hl * 8;
  ABPack ab;
  ab.u[0] = *reinterpret_cast<const uint4*>(p);
  ab.u[1] = *reinterpret_cast<const uint4*>(p + 16);
  return ab.v;
}

// B operand, 32x16 bf16 from row-major Bt[K][N] (stride ld): lane&15 = N col; lane>>4 selects K chunks {hl*8.., hl*8+16..}.
__device__ __forceinline__ v16bf load_b32x16(const unsigned short* base, int ld, int k0, int n0, int lane) {
  const int hl = lane >> 4, c = lane & 15;
  const unsigned short* p = base + (size_t)(k0 + hl * 8) * ld + n0 + c;
  union { unsigned short s[16]; v16bf v; } ab;
#pragma unroll
  for (int s = 0; s < 8; ++s) {
    ab.s[s]     = p[(size_t)s * ld];
    ab.s[s + 8] = p[(size_t)(s + 16) * ld];
  }
  return ab.v;
}

__device__ __forceinline__ v8f wmma_bf16(v16bf a, v16bf b, v8f c) {
  return __builtin_amdgcn_wmma_f32_16x16x32_bf16(false, a, false, b, (short)0, c, false, false);
}

// 16-byte global -> LDS copy: async-to-LDS path if available, else load+ds_store.
__device__ __forceinline__ void async_cp16(const unsigned short* gsrc, unsigned short* ldst) {
#if MHA_ASYNC_LDS
  __builtin_amdgcn_global_load_async_to_lds_b128(
      (gbl_v4i*)(unsigned long long)(size_t)gsrc,   // global src (addrspace 1)
      (lds_v4i*)(unsigned int)(size_t)ldst,         // LDS dst: low 32 bits of generic addr
      0, 0);
#else
  *reinterpret_cast<uint4*>(ldst) = *reinterpret_cast<const uint4*>(gsrc);
#endif
}
__device__ __forceinline__ void wait_async_le4() {
#if MHA_ASYNC_LDS
  __builtin_amdgcn_s_wait_asynccnt(4);
#endif
}
__device__ __forceinline__ void wait_async_le0() {
#if MHA_ASYNC_LDS
  __builtin_amdgcn_s_wait_asynccnt(0);
#endif
}

// ---------------- conversion kernels ----------------

__global__ void cvt_f32_bf16_kernel(const float* __restrict__ in, unsigned short* __restrict__ out, int n) {
  int i = blockIdx.x * blockDim.x + threadIdx.x;
  if (i < n) out[i] = f2bf(in[i]);
}

// wt[k][n] = w[n][k]  (both D x D)
__global__ void transpose_w_kernel(const float* __restrict__ w, unsigned short* __restrict__ wt) {
  using namespace mha;
  int i = blockIdx.x * blockDim.x + threadIdx.x;
  if (i < D_ * D_) {
    int k = i >> 10, n = i & (D_ - 1);
    wt[i] = f2bf(w[(size_t)n * D_ + k]);
  }
}

// ---------------- unified WMMA GEMM: out = A (M x D) @ Bt (D x D) ----------------
// 4 waves/WG, 64x128 tile, B panel double-buffered in LDS via async-to-LDS.
// mode 0: bf16 out [B,H,S,DK]; mode 1: bf16 out [B,H,DK,S]; mode 2: f32 out [M,D] + bias.
__global__ __launch_bounds__(128) void gemm_wmma_kernel(const unsigned short* __restrict__ amat,
                                                        const unsigned short* __restrict__ wt,
                                                        unsigned short* __restrict__ outb,
                                                        float* __restrict__ fout,
                                                        const float* __restrict__ bias,
                                                        int mode) {
  using namespace mha;
  __shared__ __align__(16) unsigned short bsm[2][BK * LDB];

  const int tid  = threadIdx.x;
  const int wave = tid >> 5;
  const int lane = tid & 31;
  const int m0 = blockIdx.x * WGM;
  const int n0 = blockIdx.y * WGN;
  const unsigned short* arow = amat + (size_t)(m0 + wave * 16) * D_;

  // stage one 32 x 128 bf16 B panel (8 KB = 512 x 16B packets, 4 per thread)
  auto stage = [&](int k0, int buf) {
#pragma unroll
    for (int j = 0; j < 4; ++j) {
      int i = tid + 128 * j;
      int row = i >> 4;            // 16 packets per 256B row
      int c8  = (i & 15) * 8;      // elem offset in row
      async_cp16(wt + (size_t)(k0 + row) * D_ + n0 + c8, &bsm[buf][row * LDB + c8]);
    }
  };

  v8f acc[8] = {};
  stage(0, 0);
  int cur = 0;
  for (int k0 = 0; k0 < D_; k0 += BK) {
    const bool has_next = (k0 + BK) < D_;
    if (has_next) {
      stage(k0 + BK, cur ^ 1);
      wait_async_le4();            // current panel done, next panel stays in flight
    } else {
      wait_async_le0();
    }
    __syncthreads();

    if (has_next) __builtin_prefetch(arow + k0 + BK, 0, 1);
    v16bf a = load_a16x32(arow, D_, k0, lane);
#pragma unroll
    for (int t = 0; t < 8; ++t) {
      v16bf b = load_b32x16(&bsm[cur][0], LDB, 0, 16 * t, lane);
      acc[t] = wmma_bf16(a, b, acc[t]);
    }
    __syncthreads();               // all reads of bsm[cur] done before it is overwritten
    cur ^= 1;
  }

  const int hl = lane >> 4, nl = lane & 15;
  const int mbase = m0 + wave * 16;
#pragma unroll
  for (int t = 0; t < 8; ++t) {
    int n = n0 + 16 * t + nl;
    if (mode == 2) {
      float bi = bias[n];
#pragma unroll
      for (int r = 0; r < 8; ++r) {
        int m = mbase + r + 8 * hl;
        fout[(size_t)m * D_ + n] = acc[t][r] + bi;
      }
    } else if (mode == 1) {
      int h = n >> 6, dd = n & 63;
#pragma unroll
      for (int r = 0; r < 8; ++r) {
        int m = mbase + r + 8 * hl;
        int b = m >> 11, s = m & (S_ - 1);
        outb[((size_t)(b * H_ + h) * DK_ + dd) * S_ + s] = f2bf(acc[t][r]);   // Kt[B,H,DK,S]
      }
    } else {
      int h = n >> 6, dd = n & 63;
#pragma unroll
      for (int r = 0; r < 8; ++r) {
        int m = mbase + r + 8 * hl;
        int b = m >> 11, s = m & (S_ - 1);
        outb[((size_t)(b * H_ + h) * S_ + s) * DK_ + dd] = f2bf(acc[t][r]);   // [B,H,S,DK]
      }
    }
  }
}

// ---------------- flash attention (causal, online softmax) ----------------
__global__ __launch_bounds__(32) void attn_kernel(const unsigned short* __restrict__ Q,
                                                  const unsigned short* __restrict__ Kt,
                                                  const unsigned short* __restrict__ V,
                                                  unsigned short* __restrict__ Cmb) {
  using namespace mha;
  __shared__ __align__(16) unsigned short lds_p[16 * 32];  // P tile staging (C-layout -> A-layout)

  const int lane = threadIdx.x;
  const int bid  = blockIdx.x;
  const int qt   = bid & (S_ / 16 - 1);
  const int bh   = bid >> 7;              // b*H + h
  const int b    = bh >> 4, h = bh & 15;
  const int q0   = qt * 16;
  const int hl = lane >> 4, nl = lane & 15;

  const unsigned short* Qb  = Q  + (size_t)bh * S_ * DK_;   // [S][64]
  const unsigned short* Ktb = Kt + (size_t)bh * DK_ * S_;   // [64][S]
  const unsigned short* Vb  = V  + (size_t)bh * S_ * DK_;   // [S][64]

  v16bf qa0 = load_a16x32(Qb + (size_t)q0 * DK_, DK_, 0, lane);
  v16bf qa1 = load_a16x32(Qb + (size_t)q0 * DK_, DK_, 32, lane);

  v8f acc[4] = {};
  float mrun[8], lrun[8];
#pragma unroll
  for (int r = 0; r < 8; ++r) { mrun[r] = -3.0e38f; lrun[r] = 0.0f; }

  for (int j0 = 0; j0 <= q0 + 15; j0 += 32) {
    v8f sc[2] = {};
#pragma unroll
    for (int t = 0; t < 2; ++t) {
      sc[t] = wmma_bf16(qa0, load_b32x16(Ktb, S_, 0,  j0 + 16 * t, lane), sc[t]);
      sc[t] = wmma_bf16(qa1, load_b32x16(Ktb, S_, 32, j0 + 16 * t, lane), sc[t]);
    }

#pragma unroll
    for (int t = 0; t < 2; ++t)
#pragma unroll
      for (int r = 0; r < 8; ++r) {
        int key  = j0 + 16 * t + nl;
        int qrow = q0 + r + 8 * hl;
        float v = sc[t][r] * 0.125f;               // 1/sqrt(64)
        sc[t][r] = (key <= qrow) ? v : -3.0e38f;
      }

    float mnew[8], scl[8];
#pragma unroll
    for (int r = 0; r < 8; ++r) {
      float v = fmaxf(sc[0][r], sc[1][r]);
      v = fmaxf(v, __shfl_xor(v, 1, 32));
      v = fmaxf(v, __shfl_xor(v, 2, 32));
      v = fmaxf(v, __shfl_xor(v, 4, 32));
      v = fmaxf(v, __shfl_xor(v, 8, 32));
      mnew[r] = fmaxf(mrun[r], v);
      scl[r]  = __expf(mrun[r] - mnew[r]);
      mrun[r] = mnew[r];
    }

#pragma unroll
    for (int t = 0; t < 2; ++t)
#pragma unroll
      for (int r = 0; r < 8; ++r)
        sc[t][r] = __expf(sc[t][r] - mnew[r]);

#pragma unroll
    for (int r = 0; r < 8; ++r) {
      float s2 = sc[0][r] + sc[1][r];
      s2 += __shfl_xor(s2, 1, 32);
      s2 += __shfl_xor(s2, 2, 32);
      s2 += __shfl_xor(s2, 4, 32);
      s2 += __shfl_xor(s2, 8, 32);
      lrun[r] = lrun[r] * scl[r] + s2;
#pragma unroll
      for (int t = 0; t < 4; ++t) acc[t][r] *= scl[r];
    }

    __syncthreads();
#pragma unroll
    for (int t = 0; t < 2; ++t)
#pragma unroll
      for (int r = 0; r < 8; ++r)
        lds_p[(r + 8 * hl) * 32 + 16 * t + nl] = f2bf(sc[t][r]);
    __syncthreads();
    v16bf pa = load_a16x32(lds_p, 32, 0, lane);

#pragma unroll
    for (int t = 0; t < 4; ++t) {
      v16bf vb = load_b32x16(Vb, DK_, j0, 16 * t, lane);
      acc[t] = wmma_bf16(pa, vb, acc[t]);
    }
  }

#pragma unroll
  for (int r = 0; r < 8; ++r) {
    float inv = 1.0f / lrun[r];
    int s = q0 + r + 8 * hl;
#pragma unroll
    for (int t = 0; t < 4; ++t) {
      int dd = 16 * t + nl;
      Cmb[(size_t)(b * S_ + s) * D_ + h * DK_ + dd] = f2bf(acc[t][r] * inv);
    }
  }
}

// ---------------- host launch ----------------

extern "C" void kernel_launch(void* const* d_in, const int* in_sizes, int n_in,
                              void* d_out, int out_size, void* d_ws, size_t ws_size,
                              hipStream_t stream) {
  using namespace mha;
  (void)in_sizes; (void)n_in; (void)out_size; (void)ws_size;

  const float* x  = (const float*)d_in[0];
  const float* Wq = (const float*)d_in[1];
  const float* Wk = (const float*)d_in[2];
  const float* Wv = (const float*)d_in[3];
  const float* Wo = (const float*)d_in[4];
  const float* bo = (const float*)d_in[5];
  float* out = (float*)d_out;

  char* ws = (char*)d_ws;
  size_t off = 0;
  auto take = [&](size_t bytes) -> unsigned short* {
    unsigned short* p = (unsigned short*)(ws + off);
    off += (bytes + 255) & ~size_t(255);
    return p;
  };
  const size_t szXD = (size_t)M_ * D_ * sizeof(unsigned short);   // 16 MB (== B*H*S*DK)
  const size_t szW  = (size_t)D_ * D_ * sizeof(unsigned short);   // 2 MB
  unsigned short* xb  = take(szXD);
  unsigned short* wqt = take(szW);
  unsigned short* wkt = take(szW);
  unsigned short* wvt = take(szW);
  unsigned short* wot = take(szW);
  unsigned short* Qm  = take(szXD);   // [B,H,S,DK]
  unsigned short* Ktm = take(szXD);   // [B,H,DK,S]
  unsigned short* Vm  = take(szXD);   // [B,H,S,DK]
  unsigned short* Cmb = take(szXD);   // [B,S,D]

  const int nX = M_ * D_;
  cvt_f32_bf16_kernel<<<(nX + 255) / 256, 256, 0, stream>>>(x, xb, nX);

  const int nW = D_ * D_;
  transpose_w_kernel<<<(nW + 255) / 256, 256, 0, stream>>>(Wq, wqt);
  transpose_w_kernel<<<(nW + 255) / 256, 256, 0, stream>>>(Wk, wkt);
  transpose_w_kernel<<<(nW + 255) / 256, 256, 0, stream>>>(Wv, wvt);
  transpose_w_kernel<<<(nW + 255) / 256, 256, 0, stream>>>(Wo, wot);

  dim3 gg(M_ / WGM, D_ / WGN);   // 128 x 8 workgroups, 128 threads each
  gemm_wmma_kernel<<<gg, 128, 0, stream>>>(xb, wqt, Qm,  nullptr, nullptr, 0);
  gemm_wmma_kernel<<<gg, 128, 0, stream>>>(xb, wkt, Ktm, nullptr, nullptr, 1);
  gemm_wmma_kernel<<<gg, 128, 0, stream>>>(xb, wvt, Vm,  nullptr, nullptr, 0);

  attn_kernel<<<B_ * H_ * (S_ / 16), 32, 0, stream>>>(Qm, Ktm, Vm, Cmb);

  gemm_wmma_kernel<<<gg, 128, 0, stream>>>(Cmb, wot, nullptr, out, bo, 2);
}
